// FrozenPocketEncoder_35957466202614
// MI455X (gfx1250) — compile-verified, matrix operand order checked by reading, not run on hardware
//
#include <hip/hip_runtime.h>
#include <hip/hip_bf16.h>
#include <math.h>

// ---------------------------------------------------------------------------
// EGNN forward for MI455X (gfx1250, wave32, WMMA bf16 16x16x32, f32 accum).
// Roofline: ~175 GFLOP of GEMM vs ~1.3 GB of gather/scatter traffic; bf16
// activations halve the dominant h[row]/h[col] gather bytes and map straight
// onto v_wmma_f32_16x16x32_bf16. Edge MLP fully fused (m never hits HBM),
// segment_sum via global_atomic_add_f32 (agg fits in 192MB L2).
// This revision: persistent tile loops (amortize LDS weight staging) and
// GLOBAL_LOAD_ASYNC_TO_LDS_B128 for the hot h-gather (ASYNCcnt path).
// ---------------------------------------------------------------------------

typedef __bf16 bf16;
typedef __attribute__((ext_vector_type(16))) __bf16 v16bf;
typedef __attribute__((ext_vector_type(8)))  __bf16 bf16x8;
typedef __attribute__((ext_vector_type(8)))  float  v8f;

#define N_NODES  50000
#define N_EDGES  600000
#define IN_DIM   64
#define HID      128
#define EDGE_DIM 16
#define OUT_DIM  128

// Padded LDS row strides (elements) to avoid 16-way DS bank conflicts.
#define KP1 296   // edge1 K (273 -> pad 288 for WMMA, 296 for banks)
#define KP2 136   // 128-K matrices
#define KPN 264   // node1 K=256
#define KPE 72    // embed K=64

static __device__ __forceinline__ v8f wmma_bf16(v16bf a, v16bf b, v8f c) {
  return __builtin_amdgcn_wmma_f32_16x16x32_bf16(false, a, false, b, (short)0, c,
                                                 false, false);
}

// Load a 16x32 bf16 A/B fragment from LDS. Per ISA layout: lanes 0-15 hold
// row=(lane&15) with K {k0..k0+7, k0+16..k0+23}; lanes 16-31 hold the same
// rows with K {k0+8..k0+15, k0+24..k0+31}. Two ds_load_b128 per fragment.
static __device__ __forceinline__ v16bf ld_frag(const bf16* base, int stride,
                                                int row, int k0, int lane) {
  const bf16* p = base + row * stride + k0 + ((lane >> 4) << 3);
  bf16x8 lo = *(const bf16x8*)p;
  bf16x8 hi = *(const bf16x8*)(p + 16);
  v16bf r;
#pragma unroll
  for (int i = 0; i < 8; ++i) { r[i] = lo[i]; r[i + 8] = hi[i]; }
  return r;
}

static __device__ __forceinline__ float siluf(float x) { return x / (1.f + __expf(-x)); }
static __device__ __forceinline__ float sigmf(float x) { return 1.f / (1.f + __expf(-x)); }

// Sum across each 16-lane half of the wave (matches C/D layout halves).
static __device__ __forceinline__ float red16(float v) {
  v += __shfl_xor(v, 1);
  v += __shfl_xor(v, 2);
  v += __shfl_xor(v, 4);
  v += __shfl_xor(v, 8);
  return v;
}

// Intra-wave LDS producer->consumer fence (LDS ops are in-order per wave;
// just drain DScnt — no block barrier needed).
static __device__ __forceinline__ void wave_lds_fence() {
  asm volatile("s_wait_dscnt 0" ::: "memory");
}

// Async global->LDS copy (16B per lane), tracked with ASYNCcnt. VDST is the
// LDS byte offset within this workgroup's LDS allocation (dynamic LDS starts
// at 0 here: no static __shared__ in these kernels).
static __device__ __forceinline__ void async_gather_b128(unsigned lds_byte,
                                                         const void* gsrc) {
  unsigned long long ga = (unsigned long long)(size_t)gsrc;
  asm volatile("global_load_async_to_lds_b128 %0, %1, off"
               :: "v"(lds_byte), "v"(ga) : "memory");
}
static __device__ __forceinline__ void wave_async_fence() {
  asm volatile("s_wait_asynccnt 0" ::: "memory");
}

// ---------------------------------------------------------------------------
// Utility kernels
// ---------------------------------------------------------------------------
__global__ void zero_f32(float* p, long n) {
  long i = (long)blockIdx.x * blockDim.x + threadIdx.x;
  long stride = (long)gridDim.x * blockDim.x;
  for (; i < n; i += stride) p[i] = 0.f;
}

__global__ void count_rows(const int* __restrict__ eidx, float* __restrict__ cnt) {
  int i = blockIdx.x * blockDim.x + threadIdx.x;
  if (i < N_EDGES) unsafeAtomicAdd(&cnt[eidx[i]], 1.f);
}

// ---------------------------------------------------------------------------
// Embed: h = h_in @ W(64,128) + b ; also emit bf16 copy for edge gathers.
// ---------------------------------------------------------------------------
#define EMB_SMEM 37376
__global__ __launch_bounds__(256) void embed_kernel(
    const float* __restrict__ hin, const float* __restrict__ wg,
    const float* __restrict__ bg, float* __restrict__ hout,
    bf16* __restrict__ hbf) {
  extern __shared__ char smem[];
  bf16*  we = (bf16*)(smem);            // [128][KPE]
  bf16*  Am = (bf16*)(smem + 18432);    // [8][16][KPE]
  float* be = (float*)(smem + 36864);

  int tid = threadIdx.x;
  for (int i = tid; i < 128 * KPE; i += 256) {
    int n = i / KPE, k = i % KPE;
    we[i] = (k < IN_DIM) ? (bf16)wg[k * 128 + n] : (bf16)0.f;
  }
  if (tid < 128) be[tid] = bg[tid];
  __syncthreads();

  int wave = tid >> 5, lane = tid & 31;
  int nl = lane & 15, mh = (lane >> 4) << 3;
  bf16* A = Am + wave * 16 * KPE;

  for (int tile = blockIdx.x * 8 + wave; tile < N_NODES / 16;
       tile += gridDim.x * 8) {
    int n0 = tile * 16;

    for (int i = lane; i < 256; i += 32) {      // 16 nodes x 16 float4 chunks
      int m = i >> 4, kc = (i & 15) << 2;
      float4 v = *(const float4*)(hin + (long)(n0 + m) * IN_DIM + kc);
      bf16* d = A + m * KPE + kc;
      d[0] = (bf16)v.x; d[1] = (bf16)v.y; d[2] = (bf16)v.z; d[3] = (bf16)v.w;
    }
    wave_lds_fence();

    v8f acc[8];
#pragma unroll
    for (int nt = 0; nt < 8; ++nt)
#pragma unroll
      for (int i = 0; i < 8; ++i) acc[nt][i] = 0.f;

#pragma unroll
    for (int kt = 0; kt < 2; ++kt) {
      v16bf af = ld_frag(A, KPE, nl, kt * 32, lane);
#pragma unroll
      for (int nt = 0; nt < 8; ++nt) {
        v16bf bf_ = ld_frag(we, KPE, nt * 16 + nl, kt * 32, lane);
        acc[nt] = wmma_bf16(af, bf_, acc[nt]);
      }
    }
#pragma unroll
    for (int nt = 0; nt < 8; ++nt) {
      int N = nt * 16 + nl;
      float bias = be[N];
#pragma unroll
      for (int r = 0; r < 8; ++r) {
        float v = acc[nt][r] + bias;
        long idx = (long)(n0 + mh + r) * HID + N;
        hout[idx] = v;
        hbf[idx] = (bf16)v;
      }
    }
  }
}

// ---------------------------------------------------------------------------
// Fused per-edge kernel: edge MLP + attention + agg scatter + coord head.
// One wave per 16 edges per iteration; weights (bf16, transposed) in LDS.
// ---------------------------------------------------------------------------
#define EDGE_AM_OFF 145408u
#define EDGE_SMEM 227344
__global__ __launch_bounds__(256) void edge_kernel(
    const int* __restrict__ eidx, const float* __restrict__ eattr,
    const bf16* __restrict__ hbf, const float* __restrict__ xin,
    float* __restrict__ agg, float* __restrict__ tsum,
    const float* __restrict__ w1g, const float* __restrict__ b1g,
    const float* __restrict__ w2g, const float* __restrict__ b2g,
    const float* __restrict__ awg, const float* __restrict__ abg,
    const float* __restrict__ c1wg, const float* __restrict__ c1bg,
    const float* __restrict__ c2wg) {
  extern __shared__ char smem[];
  bf16*  w1   = (bf16*)(smem);             // [128][KP1]  (edge1^T)
  bf16*  w2   = (bf16*)(smem + 75776);     // [128][KP2]  (edge2^T)
  bf16*  wc1  = (bf16*)(smem + 110592);    // [128][KP2]  (coord1^T)
  bf16*  Am   = (bf16*)(smem + EDGE_AM_OFF); // [8][16][KP1] A staging / reuse
  float* b1   = (float*)(smem + 221184);
  float* b2   = (float*)(smem + 221696);
  float* bc1  = (float*)(smem + 222208);
  float* aw   = (float*)(smem + 222720);
  float* c2w  = (float*)(smem + 223232);
  float* nd   = (float*)(smem + 223744);   // [8][16][4]: norm_diff.xyz, radial
  int*   rows = (int*)  (smem + 225792);   // [8][16]
  int*   cols = (int*)  (smem + 226304);   // [8][16]
  float* wsh  = (float*)(smem + 226816);   // [8][16] coord weights
  float* attb = (float*)(smem + 227328);

  int tid = threadIdx.x;
  for (int i = tid; i < 128 * KP1; i += 256) {
    int n = i / KP1, k = i % KP1;
    w1[i] = (k < 2 * HID + 1 + EDGE_DIM) ? (bf16)w1g[k * 128 + n] : (bf16)0.f;
  }
  for (int i = tid; i < 128 * KP2; i += 256) {
    int n = i / KP2, k = i % KP2;
    bf16 v2 = (k < 128) ? (bf16)w2g[k * 128 + n] : (bf16)0.f;
    bf16 vc = (k < 128) ? (bf16)c1wg[k * 128 + n] : (bf16)0.f;
    w2[i] = v2;
    wc1[i] = vc;
  }
  if (tid < 128) {
    b1[tid] = b1g[tid]; b2[tid] = b2g[tid]; bc1[tid] = c1bg[tid];
    aw[tid] = awg[tid]; c2w[tid] = c2wg[tid];
  }
  if (tid == 0) attb[0] = abg[0];
  __syncthreads();

  int wave = tid >> 5, lane = tid & 31;
  int nl = lane & 15, mh = (lane >> 4) << 3;
  bf16* A = Am + wave * 16 * KP1;
  const unsigned Aoff = EDGE_AM_OFF + (unsigned)(wave * 16 * KP1) * 2u;

  for (int tile = blockIdx.x * 8 + wave; tile < N_EDGES / 16;
       tile += gridDim.x * 8) {                // E % 16 == 0: tiles are full
    int e0 = tile * 16;

    // --- stage tail of A (radial | edge_attr | pad) + geometry, 1 lane/edge
    if (lane < 16) {
      int e = e0 + lane;
      int r = eidx[e], c = eidx[N_EDGES + e];
      rows[wave * 16 + lane] = r;
      cols[wave * 16 + lane] = c;
      float dx = xin[3 * r + 0] - xin[3 * c + 0];
      float dy = xin[3 * r + 1] - xin[3 * c + 1];
      float dz = xin[3 * r + 2] - xin[3 * c + 2];
      float rad = dx * dx + dy * dy + dz * dz;
      float inv = 1.f / (sqrtf(rad) + 1.f);
      float* n4 = nd + (wave * 16 + lane) * 4;
      n4[0] = dx * inv; n4[1] = dy * inv; n4[2] = dz * inv; n4[3] = rad;
      bf16* a = A + lane * KP1;
      a[256] = (bf16)rad;
#pragma unroll
      for (int t = 0; t < EDGE_DIM; ++t)
        a[257 + t] = (bf16)eattr[(long)e * EDGE_DIM + t];
#pragma unroll
      for (int t = 273; t < KP1; ++t) a[t] = (bf16)0.f;
    }
    wave_lds_fence();

    // --- gather hi|hj straight into LDS via async tensor path (ASYNCcnt) ---
    for (int i = lane; i < 256; i += 32) {     // 16 edges x 16 segments (32B)
      int e = i >> 4, seg = i & 15;
      int node = (seg < 8) ? rows[wave * 16 + e] : cols[wave * 16 + e];
      const bf16* src = hbf + (long)node * HID + (seg & 7) * 16;
      unsigned dstoff = Aoff +
          (unsigned)(e * KP1 + ((seg < 8) ? 0 : HID) + (seg & 7) * 16) * 2u;
      async_gather_b128(dstoff, src);
      async_gather_b128(dstoff + 16u, src + 8);
    }
    wave_async_fence();

    v8f acc[8];

    // --- GEMM1: [16,288] x [288,128] ---
#pragma unroll
    for (int nt = 0; nt < 8; ++nt)
#pragma unroll
      for (int i = 0; i < 8; ++i) acc[nt][i] = 0.f;
    for (int kt = 0; kt < 9; ++kt) {
      v16bf af = ld_frag(A, KP1, nl, kt * 32, lane);
#pragma unroll
      for (int nt = 0; nt < 8; ++nt) {
        v16bf bf_ = ld_frag(w1, KP1, nt * 16 + nl, kt * 32, lane);
        acc[nt] = wmma_bf16(af, bf_, acc[nt]);
      }
    }
    // bias + silu -> restage as bf16 A (cols < 128 of same padded region)
#pragma unroll
    for (int nt = 0; nt < 8; ++nt) {
      float bias = b1[nt * 16 + nl];
#pragma unroll
      for (int r = 0; r < 8; ++r) {
        float v = siluf(acc[nt][r] + bias);
        A[(r + mh) * KP1 + nt * 16 + nl] = (bf16)v;
      }
    }
    wave_lds_fence();

    // --- GEMM2: [16,128] x [128,128] ---
#pragma unroll
    for (int nt = 0; nt < 8; ++nt)
#pragma unroll
      for (int i = 0; i < 8; ++i) acc[nt][i] = 0.f;
#pragma unroll
    for (int kt = 0; kt < 4; ++kt) {
      v16bf af = ld_frag(A, KP1, nl, kt * 32, lane);
#pragma unroll
      for (int nt = 0; nt < 8; ++nt) {
        v16bf bf_ = ld_frag(w2, KP2, nt * 16 + nl, kt * 32, lane);
        acc[nt] = wmma_bf16(af, bf_, acc[nt]);
      }
    }
#pragma unroll
    for (int nt = 0; nt < 8; ++nt) {
      float bias = b2[nt * 16 + nl];
#pragma unroll
      for (int r = 0; r < 8; ++r) acc[nt][r] = siluf(acc[nt][r] + bias);
    }

    // --- attention gate: sigmoid(m . aw + ab), per edge ---
    float p[8];
#pragma unroll
    for (int r = 0; r < 8; ++r) p[r] = 0.f;
#pragma unroll
    for (int nt = 0; nt < 8; ++nt) {
      float a_ = aw[nt * 16 + nl];
#pragma unroll
      for (int r = 0; r < 8; ++r) p[r] += acc[nt][r] * a_;
    }
    float ab = attb[0];
#pragma unroll
    for (int r = 0; r < 8; ++r) p[r] = sigmf(red16(p[r]) + ab);
#pragma unroll
    for (int nt = 0; nt < 8; ++nt)
#pragma unroll
      for (int r = 0; r < 8; ++r) acc[nt][r] *= p[r];

    // --- scatter m into agg (segment_sum) + restage for coord head ---
    int rows8[8];
#pragma unroll
    for (int r = 0; r < 8; ++r) rows8[r] = rows[wave * 16 + mh + r];
#pragma unroll
    for (int nt = 0; nt < 8; ++nt) {
      int N = nt * 16 + nl;
#pragma unroll
      for (int r = 0; r < 8; ++r) {
        unsafeAtomicAdd(&agg[(long)rows8[r] * HID + N], acc[nt][r]);
        A[(r + mh) * KP1 + N] = (bf16)acc[nt][r];
      }
    }
    wave_lds_fence();

    // --- coord head: silu(coord1(m)) . coord2 ---
#pragma unroll
    for (int nt = 0; nt < 8; ++nt)
#pragma unroll
      for (int i = 0; i < 8; ++i) acc[nt][i] = 0.f;
#pragma unroll
    for (int kt = 0; kt < 4; ++kt) {
      v16bf af = ld_frag(A, KP1, nl, kt * 32, lane);
#pragma unroll
      for (int nt = 0; nt < 8; ++nt) {
        v16bf bf_ = ld_frag(wc1, KP2, nt * 16 + nl, kt * 32, lane);
        acc[nt] = wmma_bf16(af, bf_, acc[nt]);
      }
    }
    float q[8];
#pragma unroll
    for (int r = 0; r < 8; ++r) q[r] = 0.f;
#pragma unroll
    for (int nt = 0; nt < 8; ++nt) {
      float bias = bc1[nt * 16 + nl];
      float cw = c2w[nt * 16 + nl];
#pragma unroll
      for (int r = 0; r < 8; ++r) q[r] += siluf(acc[nt][r] + bias) * cw;
    }
#pragma unroll
    for (int r = 0; r < 8; ++r) q[r] = red16(q[r]);
    if (nl == 0) {
#pragma unroll
      for (int r = 0; r < 8; ++r) wsh[wave * 16 + mh + r] = q[r];
    }
    wave_lds_fence();

    if (lane < 16) {
      int le = wave * 16 + lane;
      float wv = wsh[le];
      int r = rows[le];
      float* n4 = nd + le * 4;
      unsafeAtomicAdd(&tsum[(long)r * 4 + 0], n4[0] * wv);
      unsafeAtomicAdd(&tsum[(long)r * 4 + 1], n4[1] * wv);
      unsafeAtomicAdd(&tsum[(long)r * 4 + 2], n4[2] * wv);
    }
  }
}

// ---------------------------------------------------------------------------
// Fused node update: h_out = LN(h + node2(silu(node1([h|agg])))) ; x update.
// ---------------------------------------------------------------------------
#define NODE_SMEM 172032
__global__ __launch_bounds__(256) void node_kernel(
    const float* __restrict__ hin, const float* __restrict__ aggp,
    const float* __restrict__ tsum, const float* __restrict__ cnt,
    const float* __restrict__ xin, float* __restrict__ xout,
    float* __restrict__ hout, bf16* __restrict__ hbf,
    const float* __restrict__ wn1g, const float* __restrict__ bn1g,
    const float* __restrict__ wn2g, const float* __restrict__ bn2g,
    const float* __restrict__ lng, const float* __restrict__ lnb) {
  extern __shared__ char smem[];
  bf16*  wn1 = (bf16*)(smem);            // [128][KPN] (node1^T)
  bf16*  wn2 = (bf16*)(smem + 67584);    // [128][KP2] (node2^T)
  bf16*  Am  = (bf16*)(smem + 102400);   // [8][16][KPN]
  float* bn1 = (float*)(smem + 169984);
  float* bn2 = (float*)(smem + 170496);
  float* lg  = (float*)(smem + 171008);
  float* lb  = (float*)(smem + 171520);

  int tid = threadIdx.x;
  for (int i = tid; i < 128 * KPN; i += 256) {
    int n = i / KPN, k = i % KPN;
    wn1[i] = (k < 256) ? (bf16)wn1g[k * 128 + n] : (bf16)0.f;
  }
  for (int i = tid; i < 128 * KP2; i += 256) {
    int n = i / KP2, k = i % KP2;
    wn2[i] = (k < 128) ? (bf16)wn2g[k * 128 + n] : (bf16)0.f;
  }
  if (tid < 128) { bn1[tid] = bn1g[tid]; bn2[tid] = bn2g[tid]; lg[tid] = lng[tid]; lb[tid] = lnb[tid]; }
  __syncthreads();

  int wave = tid >> 5, lane = tid & 31;
  int nl = lane & 15, mh = (lane >> 4) << 3;
  bf16* A = Am + wave * 16 * KPN;

  for (int tile = blockIdx.x * 8 + wave; tile < N_NODES / 16;
       tile += gridDim.x * 8) {                // N % 16 == 0
    int n0 = tile * 16;

    // stage [h | agg] as bf16
    for (int i = lane; i < 1024; i += 32) {    // 16 nodes x 64 float4 chunks
      int m = i >> 6, kc = (i & 63) << 2;
      const float* src = (kc < 128) ? (hin + (long)(n0 + m) * HID + kc)
                                    : (aggp + (long)(n0 + m) * HID + (kc - 128));
      float4 v = *(const float4*)src;
      bf16* d = A + m * KPN + kc;
      d[0] = (bf16)v.x; d[1] = (bf16)v.y; d[2] = (bf16)v.z; d[3] = (bf16)v.w;
    }
    wave_lds_fence();

    v8f acc[8];
#pragma unroll
    for (int nt = 0; nt < 8; ++nt)
#pragma unroll
      for (int i = 0; i < 8; ++i) acc[nt][i] = 0.f;
#pragma unroll
    for (int kt = 0; kt < 8; ++kt) {
      v16bf af = ld_frag(A, KPN, nl, kt * 32, lane);
#pragma unroll
      for (int nt = 0; nt < 8; ++nt) {
        v16bf bf_ = ld_frag(wn1, KPN, nt * 16 + nl, kt * 32, lane);
        acc[nt] = wmma_bf16(af, bf_, acc[nt]);
      }
    }
#pragma unroll
    for (int nt = 0; nt < 8; ++nt) {
      float bias = bn1[nt * 16 + nl];
#pragma unroll
      for (int r = 0; r < 8; ++r) {
        float v = siluf(acc[nt][r] + bias);
        A[(r + mh) * KPN + nt * 16 + nl] = (bf16)v;
      }
    }
    wave_lds_fence();

#pragma unroll
    for (int nt = 0; nt < 8; ++nt)
#pragma unroll
      for (int i = 0; i < 8; ++i) acc[nt][i] = 0.f;
#pragma unroll
    for (int kt = 0; kt < 4; ++kt) {
      v16bf af = ld_frag(A, KPN, nl, kt * 32, lane);
#pragma unroll
      for (int nt = 0; nt < 8; ++nt) {
        v16bf bf_ = ld_frag(wn2, KP2, nt * 16 + nl, kt * 32, lane);
        acc[nt] = wmma_bf16(af, bf_, acc[nt]);
      }
    }
    // residual + bias
#pragma unroll
    for (int nt = 0; nt < 8; ++nt) {
      int N = nt * 16 + nl;
      float bias = bn2[N];
#pragma unroll
      for (int r = 0; r < 8; ++r)
        acc[nt][r] += bias + hin[(long)(n0 + mh + r) * HID + N];
    }

    // LayerNorm over the 128 features of each node (cross-lane stats)
    float s1[8], s2[8];
#pragma unroll
    for (int r = 0; r < 8; ++r) { s1[r] = 0.f; s2[r] = 0.f; }
#pragma unroll
    for (int nt = 0; nt < 8; ++nt)
#pragma unroll
      for (int r = 0; r < 8; ++r) { float v = acc[nt][r]; s1[r] += v; s2[r] += v * v; }
    float mu[8], ri[8];
#pragma unroll
    for (int r = 0; r < 8; ++r) {
      float m = red16(s1[r]) * (1.f / 128.f);
      float v = red16(s2[r]) * (1.f / 128.f) - m * m;
      mu[r] = m;
      ri[r] = rsqrtf(v + 1e-5f);
    }
#pragma unroll
    for (int nt = 0; nt < 8; ++nt) {
      int N = nt * 16 + nl;
      float g = lg[N], b = lb[N];
#pragma unroll
      for (int r = 0; r < 8; ++r) {
        float v = (acc[nt][r] - mu[r]) * ri[r] * g + b;
        long idx = (long)(n0 + mh + r) * HID + N;
        hout[idx] = v;
        hbf[idx] = (bf16)v;
      }
    }

    // coordinate mean-update
    if (lane < 16) {
      int node = n0 + lane;
      float c = fmaxf(cnt[node], 1.f);
#pragma unroll
      for (int j = 0; j < 3; ++j)
        xout[(long)node * 3 + j] = xin[(long)node * 3 + j] + tsum[(long)node * 4 + j] / c;
    }
  }
}

// ---------------------------------------------------------------------------
// Output projection: d_out = h @ W(128,128) + b
// ---------------------------------------------------------------------------
#define OUT_SMEM 70144
__global__ __launch_bounds__(256) void out_kernel(
    const float* __restrict__ hin, const float* __restrict__ wg,
    const float* __restrict__ bg, float* __restrict__ out) {
  extern __shared__ char smem[];
  bf16*  wo = (bf16*)(smem);            // [128][KP2]
  bf16*  Am = (bf16*)(smem + 34816);    // [8][16][KP2]
  float* bo = (float*)(smem + 69632);

  int tid = threadIdx.x;
  for (int i = tid; i < 128 * KP2; i += 256) {
    int n = i / KP2, k = i % KP2;
    wo[i] = (k < 128) ? (bf16)wg[k * 128 + n] : (bf16)0.f;
  }
  if (tid < 128) bo[tid] = bg[tid];
  __syncthreads();

  int wave = tid >> 5, lane = tid & 31;
  int nl = lane & 15, mh = (lane >> 4) << 3;
  bf16* A = Am + wave * 16 * KP2;

  for (int tile = blockIdx.x * 8 + wave; tile < N_NODES / 16;
       tile += gridDim.x * 8) {
    int n0 = tile * 16;

    for (int i = lane; i < 512; i += 32) {     // 16 nodes x 32 float4 chunks
      int m = i >> 5, kc = (i & 31) << 2;
      float4 v = *(const float4*)(hin + (long)(n0 + m) * HID + kc);
      bf16* d = A + m * KP2 + kc;
      d[0] = (bf16)v.x; d[1] = (bf16)v.y; d[2] = (bf16)v.z; d[3] = (bf16)v.w;
    }
    wave_lds_fence();

    v8f acc[8];
#pragma unroll
    for (int nt = 0; nt < 8; ++nt)
#pragma unroll
      for (int i = 0; i < 8; ++i) acc[nt][i] = 0.f;
#pragma unroll
    for (int kt = 0; kt < 4; ++kt) {
      v16bf af = ld_frag(A, KP2, nl, kt * 32, lane);
#pragma unroll
      for (int nt = 0; nt < 8; ++nt) {
        v16bf bf_ = ld_frag(wo, KP2, nt * 16 + nl, kt * 32, lane);
        acc[nt] = wmma_bf16(af, bf_, acc[nt]);
      }
    }
#pragma unroll
    for (int nt = 0; nt < 8; ++nt) {
      int N = nt * 16 + nl;
      float bias = bo[N];
#pragma unroll
      for (int r = 0; r < 8; ++r)
        out[(long)(n0 + mh + r) * OUT_DIM + N] = acc[nt][r] + bias;
    }
  }
}

// ---------------------------------------------------------------------------
// Host-side orchestration.
// Input leaf order (setup_inputs insertion order, params flattened depth-first
// in construction order):
//   0:h 1:x 2:edge_index 3:edge_attr 4:embed.w 5:embed.b 6:out.w 7:out.b
//   8+13*L: edge1.w,edge1.b,edge2.w,edge2.b,att.w,att.b,node1.w,node1.b,
//           node2.w,node2.b,coord1.w,coord1.b,coord2.w   (L = 0,1)
//   34:ln0.g 35:ln0.b 36:ln1.g 37:ln1.b
// ---------------------------------------------------------------------------
extern "C" void kernel_launch(void* const* d_in, const int* in_sizes, int n_in,
                              void* d_out, int out_size, void* d_ws, size_t ws_size,
                              hipStream_t stream) {
  (void)in_sizes; (void)n_in; (void)out_size; (void)ws_size;
  const float* h_in  = (const float*)d_in[0];
  const float* x_in  = (const float*)d_in[1];
  const int*   eidx  = (const int*)d_in[2];
  const float* eattr = (const float*)d_in[3];
  const float* emb_w = (const float*)d_in[4];
  const float* emb_b = (const float*)d_in[5];
  const float* out_w = (const float*)d_in[6];
  const float* out_b = (const float*)d_in[7];
  const float* lnp[2][2] = {
      {(const float*)d_in[34], (const float*)d_in[35]},
      {(const float*)d_in[36], (const float*)d_in[37]}};

  char* ws = (char*)d_ws;
  float* hA   = (float*)(ws + 0);                     // 25.6 MB
  float* hB   = (float*)(ws + 25600000);              // 25.6 MB
  bf16*  hbf  = (bf16*) (ws + 51200000);              // 12.8 MB
  float* agg  = (float*)(ws + 64000000);              // 25.6 MB
  float* tsum = (float*)(ws + 89600000);              // 0.8 MB (N x 4)
  float* cnt  = (float*)(ws + 90400000);              // 0.2 MB
  float* xA   = (float*)(ws + 90600000);              // 0.6 MB
  float* xB   = (float*)(ws + 91200000);              // 0.6 MB

  const int EDGE_BLOCKS = 1024;   // persistent: 37500 tiles / (1024*8) ~ 4.6 it
  const int NODE_BLOCKS = 391;    // 3125 tiles / 8

  // per-receiver edge counts (layer-invariant)
  zero_f32<<<256, 256, 0, stream>>>(cnt, (long)N_NODES);
  count_rows<<<(N_EDGES + 255) / 256, 256, 0, stream>>>(eidx, cnt);

  embed_kernel<<<NODE_BLOCKS, 256, EMB_SMEM, stream>>>(h_in, emb_w, emb_b, hA, hbf);

  const float* hcur = hA;
  float* hnxt = hB;
  const float* xcur = x_in;
  float* xnxt = xA;
  for (int L = 0; L < 2; ++L) {
    const float* P[13];
    for (int j = 0; j < 13; ++j) P[j] = (const float*)d_in[8 + L * 13 + j];
    // P: 0 e1w 1 e1b 2 e2w 3 e2b 4 attw 5 attb 6 n1w 7 n1b 8 n2w 9 n2b
    //    10 c1w 11 c1b 12 c2w
    zero_f32<<<512, 256, 0, stream>>>(agg, (long)N_NODES * HID);
    zero_f32<<<256, 256, 0, stream>>>(tsum, (long)N_NODES * 4);
    edge_kernel<<<EDGE_BLOCKS, 256, EDGE_SMEM, stream>>>(
        eidx, eattr, hbf, xcur, agg, tsum,
        P[0], P[1], P[2], P[3], P[4], P[5], P[10], P[11], P[12]);
    node_kernel<<<NODE_BLOCKS, 256, NODE_SMEM, stream>>>(
        hcur, agg, tsum, cnt, xcur, xnxt, hnxt, hbf,
        P[6], P[7], P[8], P[9], lnp[L][0], lnp[L][1]);
    // ping-pong
    const float* t = hcur; hcur = hnxt; hnxt = (float*)t;
    xcur = xnxt; xnxt = xB;
  }

  out_kernel<<<NODE_BLOCKS, 256, OUT_SMEM, stream>>>(hcur, out_w, out_b, (float*)d_out);
}
